// PaliGemmaWithExpertModel_23167053595058
// MI455X (gfx1250) — matrix operand order repaired and checked by reading, not compiled, not run on hardware
//
#include <hip/hip_runtime.h>

// ---------------------------------------------------------------------------
// PaliGemma-with-expert fused 2-layer forward for MI455X (gfx1250, wave32).
// All GEMMs run on v_wmma_f32_16x16x32_f16 (f16 A/B, f32 accumulate).
// Per-wave 32x64 register tile (8 WMMA / K-step); async global->LDS staging
// (GLOBAL_LOAD_ASYNC_TO_LDS_B128, ASYNCcnt) for interior f16 tiles.
// Workspace requirement: ~300 MB.
// ---------------------------------------------------------------------------

typedef __attribute__((ext_vector_type(16))) _Float16 v16h;
typedef __attribute__((ext_vector_type(8)))  float    v8f;
typedef int v4i_ __attribute__((vector_size(16)));   // matches builtin proto

// ---- model dimensions -----------------------------------------------------
constexpr int BSZ = 2, SV = 768, SE = 50, T = SV + SE;   // T = 818
constexpr int LAY = 2, H = 8, Dh = 256;
constexpr int WVd = 2048, FVd = 16384, WEd = 1024, FEd = 4096;
constexpr int Tp  = 832;                                  // T padded to 64
constexpr float EPSv   = 1e-6f;
constexpr float SCALEv = 0.0625f;                         // 256^-0.5
constexpr float THETAv = 10000.0f;

// ---- CDNA5 async global->LDS path -----------------------------------------
#if defined(__has_builtin)
#if __has_builtin(__builtin_amdgcn_global_load_async_to_lds_b128) && \
    __has_builtin(__builtin_amdgcn_s_wait_asynccnt)
#define USE_ASYNC_LDS 1
#endif
#endif
#ifndef USE_ASYNC_LDS
#define USE_ASYNC_LDS 0
#endif

#if USE_ASYNC_LDS
__device__ __forceinline__ void async_cp16(void* lds, const void* g)
{
    __builtin_amdgcn_global_load_async_to_lds_b128(
        (__attribute__((address_space(1))) v4i_*)g,
        (__attribute__((address_space(3))) v4i_*)lds,
        0, 0);
}
#endif

// ===========================================================================
// Generic tiled WMMA GEMM:  C[M,N] f32  =  A[M,K] f16  x  B
//   TRANSB=true : B is [N,K] row-major (weight / K-matrix form), dtype BT
//   TRANSB=false: B is [K,N] row-major (V-matrix form),          dtype BT
// Block: 128 threads (4 waves), BM=64 x BN=128, BK=32.
// Each wave computes a 32x64 tile: 2 A-frags x 4 B-frags = 8 WMMA per K-step.
// Batch z: offsets decomposed as (z/zdiv)*s1 + (z%zdiv)*s2 per operand.
// ===========================================================================
template <typename BT, bool TRANSB>
__global__ __launch_bounds__(128)
void gemm_wmma(const _Float16* __restrict__ Ag, const BT* __restrict__ Bg,
               float* __restrict__ Cg,
               int M, int N, int K, int lda, int ldb, int ldc,
               int zdiv, long a1, long a2, long b1, long b2, long c1, long c2)
{
    constexpr int BM = 64, BN = 128, BK = 32;
    __shared__ _Float16 As[BM][BK];
    __shared__ _Float16 Bs[BN][BK];   // Bs[n][k] = B(k0+k, bn+n)

    const int z = blockIdx.z;
    Ag += (long)(z / zdiv) * a1 + (long)(z % zdiv) * a2;
    Bg += (long)(z / zdiv) * b1 + (long)(z % zdiv) * b2;
    Cg += (long)(z / zdiv) * c1 + (long)(z % zdiv) * c2;

    const int bm = blockIdx.y * BM, bn = blockIdx.x * BN;
    const int tid = threadIdx.x, lane = tid & 31, wave = tid >> 5;
    const int wm = (wave >> 1) * 32, wn = (wave & 1) * 64;

    const bool interiorA = (bm + BM <= M);
    const bool interiorB = (bn + BN <= N);
    (void)interiorB;

    v8f acc[2][4] = {};

    for (int k0 = 0; k0 < K; k0 += BK) {
        bool usedAsync = false;
        // ---------- stage A tile ----------
        {
            const int r = tid >> 1, c0 = (tid & 1) * 16;
            const _Float16* src = Ag + (long)(bm + r) * lda + k0 + c0;
#if USE_ASYNC_LDS
            if (interiorA) {
                async_cp16(&As[r][c0],     src);
                async_cp16(&As[r][c0 + 8], src + 8);
                usedAsync = true;
            } else
#endif
            {
                if (bm + r < M) {
                    const uint4* s4 = (const uint4*)src;
                    *(uint4*)&As[r][c0]     = s4[0];
                    *(uint4*)&As[r][c0 + 8] = s4[1];
                } else {
                    const uint4 zz = {0u, 0u, 0u, 0u};
                    *(uint4*)&As[r][c0]     = zz;
                    *(uint4*)&As[r][c0 + 8] = zz;
                }
            }
        }
        // ---------- stage B tile ----------
        if (TRANSB) {
#pragma unroll
            for (int half = 0; half < 2; ++half) {
                const int n   = (tid >> 1) + 64 * half;
                const int kc0 = (tid & 1) * 16;
                const int gn  = bn + n;
                const BT* src = Bg + (long)gn * ldb + k0 + kc0;
#if USE_ASYNC_LDS
                if (sizeof(BT) == 2 && interiorB) {
                    async_cp16(&Bs[n][kc0],     src);
                    async_cp16(&Bs[n][kc0 + 8], src + 8);
                    usedAsync = true;
                } else
#endif
                {
                    if (gn < N) {
#pragma unroll
                        for (int i = 0; i < 16; ++i)
                            Bs[n][kc0 + i] = (_Float16)src[i];
                    } else {
#pragma unroll
                        for (int i = 0; i < 16; ++i)
                            Bs[n][kc0 + i] = (_Float16)0.f;
                    }
                }
            }
        } else {
            const int k = tid & 31;
#pragma unroll
            for (int half = 0; half < 2; ++half) {
                const int n0 = (tid >> 5) * 16 + 64 * half;
                const BT* src = Bg + (long)(k0 + k) * ldb + bn + n0;
#pragma unroll
                for (int i = 0; i < 16; ++i) {
                    const int gn = bn + n0 + i;
                    Bs[n0 + i][k] = (gn < N) ? (_Float16)src[i] : (_Float16)0.f;
                }
            }
        }
        // prefetch the next K tile toward the caches while we compute
        if (k0 + BK < K) {
            __builtin_prefetch(Ag + (long)(bm + (tid >> 1)) * lda + k0 + BK, 0, 0);
        }
#if USE_ASYNC_LDS
        if (usedAsync) __builtin_amdgcn_s_wait_asynccnt(0);
#else
        (void)usedAsync;
#endif
        __syncthreads();

        // ---------- fragments per documented CDNA5 WMMA layouts ----------
        v16h afr[2], bfr[4];
#pragma unroll
        for (int mi = 0; mi < 2; ++mi) {
            const int r  = wm + mi * 16 + (lane & 15);
            const int kb = (lane >> 4) * 8;
#pragma unroll
            for (int i = 0; i < 8; ++i) {
                afr[mi][i]     = As[r][kb + i];
                afr[mi][8 + i] = As[r][16 + kb + i];
            }
        }
#pragma unroll
        for (int ni = 0; ni < 4; ++ni) {
            const int n  = wn + ni * 16 + (lane & 15);
            const int kb = (lane >> 4) * 8;
#pragma unroll
            for (int i = 0; i < 8; ++i) {
                bfr[ni][i]     = Bs[n][kb + i];
                bfr[ni][8 + i] = Bs[n][16 + kb + i];
            }
        }
#pragma unroll
        for (int mi = 0; mi < 2; ++mi)
#pragma unroll
            for (int ni = 0; ni < 4; ++ni)
                acc[mi][ni] = __builtin_amdgcn_wmma_f32_16x16x32_f16(
                    false, afr[mi], false, bfr[ni],
                    (short)0, acc[mi][ni], false, false);
        __syncthreads();
    }

    // ---------- store C (M/N-edge guarded) ----------
#pragma unroll
    for (int mi = 0; mi < 2; ++mi)
#pragma unroll
        for (int ni = 0; ni < 4; ++ni) {
            const int gn = bn + wn + ni * 16 + (lane & 15);
#pragma unroll
            for (int r = 0; r < 8; ++r) {
                const int gm = bm + wm + mi * 16 + r + ((lane >> 4) << 3);
                if (gm < M && gn < N) Cg[(long)gm * ldc + gn] = acc[mi][ni][r];
            }
        }
}

// ===========================================================================
// Elementwise / reduction glue kernels
// ===========================================================================
__global__ __launch_bounds__(256)
void rmsnorm_f16(const float* __restrict__ x, const float* __restrict__ w,
                 _Float16* __restrict__ out, int W)
{
    const long row = blockIdx.x;
    const float* xr = x + row * (long)W;
    float s = 0.f;
    for (int c = threadIdx.x; c < W; c += blockDim.x) { float v = xr[c]; s += v * v; }
    __shared__ float red[8];
    for (int o = 16; o > 0; o >>= 1) s += __shfl_xor(s, o, 32);
    if ((threadIdx.x & 31) == 0) red[threadIdx.x >> 5] = s;
    __syncthreads();
    if (threadIdx.x == 0) {
        float v = 0.f;
        for (int i = 0; i < 8; ++i) v += red[i];
        red[0] = v;
    }
    __syncthreads();
    const float inv = rsqrtf(red[0] / (float)W + EPSv);
    _Float16* orow = out + row * (long)W;
    for (int c = threadIdx.x; c < W; c += blockDim.x)
        orow[c] = (_Float16)(xr[c] * inv * (1.0f + w[c]));
}

__global__ void rope_pack_q(const float* __restrict__ qv, const float* __restrict__ qe,
                            const int* __restrict__ pos, _Float16* __restrict__ Q)
{
    const long idx = blockIdx.x * (long)blockDim.x + threadIdx.x;
    const long total = (long)BSZ * H * Tp * Dh;
    if (idx >= total) return;
    const int d = (int)(idx % Dh);
    long r = idx / Dh;
    const int t = (int)(r % Tp); r /= Tp;
    const int h = (int)(r % H);
    const int b = (int)(r / H);
    if (t >= T) { Q[idx] = (_Float16)0.f; return; }
    const float p  = (float)pos[b * T + t];
    const int   dm = d & 127;
    const float ang = p * __powf(THETAv, -(float)(2 * dm) / (float)Dh);
    const float c = __cosf(ang), s = __sinf(ang);
    const float* src; long row;
    if (t < SV) { src = qv; row = (long)b * SV + t; }
    else        { src = qe; row = (long)b * SE + (t - SV); }
    const float* pr = src + row * (long)(H * Dh) + (long)h * Dh;
    const float x  = pr[d];
    const float x2 = (d < 128) ? pr[d + 128] : pr[d - 128];
    const float val = (d < 128) ? (x * c - x2 * s) : (x * c + x2 * s);
    Q[idx] = (_Float16)val;
}

__global__ void rope_pack_kv(const float* __restrict__ kv, const float* __restrict__ ke,
                             const float* __restrict__ vv, const float* __restrict__ ve,
                             const int* __restrict__ pos,
                             _Float16* __restrict__ Kb, _Float16* __restrict__ Vb)
{
    const long idx = blockIdx.x * (long)blockDim.x + threadIdx.x;
    const long total = (long)BSZ * Tp * Dh;
    if (idx >= total) return;
    const int d = (int)(idx % Dh);
    long r = idx / Dh;
    const int t = (int)(r % Tp);
    const int b = (int)(r / Tp);
    if (t >= T) { Kb[idx] = (_Float16)0.f; Vb[idx] = (_Float16)0.f; return; }
    const float p  = (float)pos[b * T + t];
    const int   dm = d & 127;
    const float ang = p * __powf(THETAv, -(float)(2 * dm) / (float)Dh);
    const float c = __cosf(ang), s = __sinf(ang);
    const float *ks, *vs; long row;
    if (t < SV) { ks = kv; vs = vv; row = (long)b * SV + t; }
    else        { ks = ke; vs = ve; row = (long)b * SE + (t - SV); }
    const float* kr = ks + row * (long)Dh;
    const float  x  = kr[d];
    const float  x2 = (d < 128) ? kr[d + 128] : kr[d - 128];
    Kb[idx] = (_Float16)((d < 128) ? (x * c - x2 * s) : (x * c + x2 * s));
    Vb[idx] = (_Float16)(vs[row * (long)Dh + d]);
}

__global__ __launch_bounds__(256)
void softmax_mask(const float* __restrict__ scores, const float* __restrict__ mask,
                  _Float16* __restrict__ att)
{
    const long row = blockIdx.x;                 // b*H*Tp + h*Tp + q
    const int  q   = (int)(row % Tp);
    const int  b   = (int)(row / ((long)Tp * H));
    const float* srow = scores + row * (long)Tp;
    _Float16*    arow = att    + row * (long)Tp;
    if (q >= T) {
        for (int c = threadIdx.x; c < Tp; c += blockDim.x) arow[c] = (_Float16)0.f;
        return;
    }
    const float* mrow = mask + ((long)b * T + q) * (long)T;
    __shared__ float red[8];
    __shared__ float rmax, rsum;
    float mx = -3.0e38f;
    for (int c = threadIdx.x; c < T; c += blockDim.x)
        mx = fmaxf(mx, srow[c] * SCALEv + mrow[c]);
    for (int o = 16; o > 0; o >>= 1) mx = fmaxf(mx, __shfl_xor(mx, o, 32));
    if ((threadIdx.x & 31) == 0) red[threadIdx.x >> 5] = mx;
    __syncthreads();
    if (threadIdx.x == 0) {
        float v = red[0];
        for (int i = 1; i < 8; ++i) v = fmaxf(v, red[i]);
        rmax = v;
    }
    __syncthreads();
    float sm = 0.f;
    for (int c = threadIdx.x; c < T; c += blockDim.x)
        sm += __expf(srow[c] * SCALEv + mrow[c] - rmax);
    for (int o = 16; o > 0; o >>= 1) sm += __shfl_xor(sm, o, 32);
    if ((threadIdx.x & 31) == 0) red[threadIdx.x >> 5] = sm;
    __syncthreads();
    if (threadIdx.x == 0) {
        float v = 0.f;
        for (int i = 0; i < 8; ++i) v += red[i];
        rsum = v;
    }
    __syncthreads();
    const float inv = 1.0f / rsum;
    for (int c = threadIdx.x; c < Tp; c += blockDim.x)
        arow[c] = (c < T) ? (_Float16)(__expf(srow[c] * SCALEv + mrow[c] - rmax) * inv)
                          : (_Float16)0.f;
}

__global__ void split_attn(const float* __restrict__ attn,
                           _Float16* __restrict__ ov, _Float16* __restrict__ oe)
{
    const long idx = blockIdx.x * (long)blockDim.x + threadIdx.x;
    const long total = (long)BSZ * T * (H * Dh);
    if (idx >= total) return;
    const int c = (int)(idx % (H * Dh));
    long r = idx / (H * Dh);
    const int t = (int)(r % T);
    const int b = (int)(r / T);
    const float v = attn[((long)b * Tp + t) * (long)(H * Dh) + c];
    if (t < SV) ov[((long)b * SV + t) * (long)(H * Dh) + c]        = (_Float16)v;
    else        oe[((long)b * SE + (t - SV)) * (long)(H * Dh) + c] = (_Float16)v;
}

__global__ void add_inplace(float* __restrict__ y, const float* __restrict__ a, long n)
{
    const long idx = blockIdx.x * (long)blockDim.x + threadIdx.x;
    if (idx < n) y[idx] += a[idx];
}

__global__ void gelu_mul_f16(const float* __restrict__ g, const float* __restrict__ u,
                             _Float16* __restrict__ out, long n)
{
    const long idx = blockIdx.x * (long)blockDim.x + threadIdx.x;
    if (idx >= n) return;
    const float x  = g[idx];
    const float x3 = x * x * x;
    const float ge = 0.5f * x * (1.0f + tanhf(0.7978845608028654f * (x + 0.044715f * x3)));
    out[idx] = (_Float16)(ge * u[idx]);
}

// ===========================================================================
// Host orchestration
// ===========================================================================
static inline size_t alignup256(size_t x) { return (x + 255) & ~(size_t)255; }

extern "C" void kernel_launch(void* const* d_in, const int* in_sizes, int n_in,
                              void* d_out, int out_size, void* d_ws, size_t ws_size,
                              hipStream_t stream)
{
    (void)in_sizes; (void)n_in; (void)out_size; (void)ws_size;

    const float* in_vlm_x = (const float*)d_in[0];
    const float* in_exp_x = (const float*)d_in[1];
    const float* mask     = (const float*)d_in[2];
    const int*   pos      = (const int*)  d_in[3];
    const float* vlm_ln1  = (const float*)d_in[4];
    const float* vlm_qw   = (const float*)d_in[5];
    const float* vlm_kw   = (const float*)d_in[6];
    const float* vlm_vw   = (const float*)d_in[7];
    const float* vlm_ow   = (const float*)d_in[8];
    const float* vlm_ln2  = (const float*)d_in[9];
    const float* vlm_gw   = (const float*)d_in[10];
    const float* vlm_uw   = (const float*)d_in[11];
    const float* vlm_dw   = (const float*)d_in[12];
    const float* exp_ln1  = (const float*)d_in[13];
    const float* exp_qw   = (const float*)d_in[14];
    const float* exp_kw   = (const float*)d_in[15];
    const float* exp_vw   = (const float*)d_in[16];
    const float* exp_ow   = (const float*)d_in[17];
    const float* exp_ln2  = (const float*)d_in[18];
    const float* exp_gw   = (const float*)d_in[19];
    const float* exp_uw   = (const float*)d_in[20];
    const float* exp_dw   = (const float*)d_in[21];

    const long nXV = (long)BSZ * SV * WVd;       // vlm residual elems
    const long nXE = (long)BSZ * SE * WEd;       // expert residual elems
    const int  MV  = BSZ * SV;                   // 1536 vlm rows
    const int  ME  = BSZ * SE;                   // 100 expert rows
    const int  HD  = H * Dh;                     // 2048

    char* ws = (char*)d_ws;
    size_t off = 0;
    auto alloc = [&](size_t bytes) { size_t o = off; off = alignup256(off + bytes); return o; };

    // persistent residual streams
    float* xv = (float*)(ws + alloc((size_t)nXV * 4));
    float* xe = (float*)(ws + alloc((size_t)nXE * 4));
    const size_t poolStart = off;

    (void)hipMemcpyAsync(xv, in_vlm_x, (size_t)nXV * 4, hipMemcpyDeviceToDevice, stream);
    (void)hipMemcpyAsync(xe, in_exp_x, (size_t)nXE * 4, hipMemcpyDeviceToDevice, stream);

    // ---- GEMM launch helpers ----
    auto gemmW = [&](const _Float16* A, const float* Wt, float* C, int M, int N, int K) {
        dim3 g((N + 127) / 128, (M + 63) / 64, 1);
        hipLaunchKernelGGL((gemm_wmma<float, true>), g, dim3(128), 0, stream,
                           A, Wt, C, M, N, K, K, K, N,
                           1, (long)0, (long)0, (long)0, (long)0, (long)0, (long)0);
    };

    auto launch1d = [&](long n) { return dim3((unsigned)((n + 255) / 256)); };

    for (int i = 0; i < LAY; ++i) {
        // ================= attention phase =================
        off = poolStart;
        _Float16* hv = (_Float16*)(ws + alloc((size_t)MV * WVd * 2));
        _Float16* he = (_Float16*)(ws + alloc((size_t)ME * WEd * 2));
        float* qv = (float*)(ws + alloc((size_t)MV * HD * 4));
        float* qe = (float*)(ws + alloc((size_t)ME * HD * 4));
        float* kvb = (float*)(ws + alloc((size_t)MV * Dh * 4));
        float* vvb = (float*)(ws + alloc((size_t)MV * Dh * 4));
        float* keb = (float*)(ws + alloc((size_t)ME * Dh * 4));
        float* veb = (float*)(ws + alloc((size_t)ME * Dh * 4));
        _Float16* Qb = (_Float16*)(ws + alloc((size_t)BSZ * H * Tp * Dh * 2));
        _Float16* Kb = (_Float16*)(ws + alloc((size_t)BSZ * Tp * Dh * 2));
        _Float16* Vb = (_Float16*)(ws + alloc((size_t)BSZ * Tp * Dh * 2));
        float* scores = (float*)(ws + alloc((size_t)BSZ * H * Tp * Tp * 4));
        _Float16* attb = (_Float16*)(ws + alloc((size_t)BSZ * H * Tp * Tp * 2));
        float* attnf = (float*)(ws + alloc((size_t)BSZ * Tp * HD * 4));
        _Float16* oiv = (_Float16*)(ws + alloc((size_t)MV * HD * 2));
        _Float16* oie = (_Float16*)(ws + alloc((size_t)ME * HD * 2));
        float* oov = (float*)(ws + alloc((size_t)MV * WVd * 4));
        float* ooe = (float*)(ws + alloc((size_t)ME * WEd * 4));

        // pre-attn RMSNorm (writes f16 for WMMA)
        hipLaunchKernelGGL(rmsnorm_f16, dim3(MV), dim3(256), 0, stream, xv, vlm_ln1 + (long)i * WVd, hv, WVd);
        hipLaunchKernelGGL(rmsnorm_f16, dim3(ME), dim3(256), 0, stream, xe, exp_ln1 + (long)i * WEd, he, WEd);

        // QKV projections
        gemmW(hv, vlm_qw + (long)i * HD * WVd, qv,  MV, HD,  WVd);
        gemmW(hv, vlm_kw + (long)i * Dh * WVd, kvb, MV, Dh,  WVd);
        gemmW(hv, vlm_vw + (long)i * Dh * WVd, vvb, MV, Dh,  WVd);
        gemmW(he, exp_qw + (long)i * HD * WEd, qe,  ME, HD,  WEd);
        gemmW(he, exp_kw + (long)i * Dh * WEd, keb, ME, Dh,  WEd);
        gemmW(he, exp_vw + (long)i * Dh * WEd, veb, ME, Dh,  WEd);

        // RoPE + concat streams + pack to f16 padded [B,H,Tp,D] / [B,Tp,D]
        hipLaunchKernelGGL(rope_pack_q, launch1d((long)BSZ * H * Tp * Dh), dim3(256), 0, stream,
                           qv, qe, pos, Qb);
        hipLaunchKernelGGL(rope_pack_kv, launch1d((long)BSZ * Tp * Dh), dim3(256), 0, stream,
                           kvb, keb, vvb, veb, pos, Kb, Vb);

        // scores = Q K^T  (per b,h; K shared across heads — GQA)
        {
            dim3 g((Tp + 127) / 128, Tp / 64, BSZ * H);
            hipLaunchKernelGGL((gemm_wmma<_Float16, true>), g, dim3(128), 0, stream,
                               Qb, Kb, scores, Tp, Tp, Dh, Dh, Dh, Tp,
                               H, (long)H * Tp * Dh, (long)Tp * Dh,
                               (long)Tp * Dh, (long)0,
                               (long)H * Tp * Tp, (long)Tp * Tp);
        }
        // softmax(scale*scores + mask), pad cols zeroed, f16 out
        hipLaunchKernelGGL(softmax_mask, dim3(BSZ * H * Tp), dim3(256), 0, stream,
                           scores, mask, attb);
        // out = att @ V  -> directly into interleaved [B,Tp,H*D] f32
        {
            dim3 g((Dh + 127) / 128, Tp / 64, BSZ * H);
            hipLaunchKernelGGL((gemm_wmma<_Float16, false>), g, dim3(128), 0, stream,
                               attb, Vb, attnf, Tp, Dh, Tp, Tp, Dh, HD,
                               H, (long)H * Tp * Tp, (long)Tp * Tp,
                               (long)Tp * Dh, (long)0,
                               (long)Tp * HD, (long)Dh);
        }
        // split back into vlm/expert rows, cast to f16 for O-proj
        hipLaunchKernelGGL(split_attn, launch1d((long)BSZ * T * HD), dim3(256), 0, stream,
                           attnf, oiv, oie);
        // O projections + residual
        gemmW(oiv, vlm_ow + (long)i * WVd * HD, oov, MV, WVd, HD);
        gemmW(oie, exp_ow + (long)i * WEd * HD, ooe, ME, WEd, HD);
        hipLaunchKernelGGL(add_inplace, launch1d(nXV), dim3(256), 0, stream, xv, oov, nXV);
        hipLaunchKernelGGL(add_inplace, launch1d(nXE), dim3(256), 0, stream, xe, ooe, nXE);

        // ================= MLP phase (reuse pool) =================
        off = poolStart;
        _Float16* h2v = (_Float16*)(ws + alloc((size_t)MV * WVd * 2));
        _Float16* h2e = (_Float16*)(ws + alloc((size_t)ME * WEd * 2));
        float* gv = (float*)(ws + alloc((size_t)MV * FVd * 4));
        float* uv = (float*)(ws + alloc((size_t)MV * FVd * 4));
        _Float16* guv = (_Float16*)(ws + alloc((size_t)MV * FVd * 2));
        float* dv = (float*)(ws + alloc((size_t)MV * WVd * 4));
        float* ge = (float*)(ws + alloc((size_t)ME * FEd * 4));
        float* ue = (float*)(ws + alloc((size_t)ME * FEd * 4));
        _Float16* gue = (_Float16*)(ws + alloc((size_t)ME * FEd * 2));
        float* de = (float*)(ws + alloc((size_t)ME * WEd * 4));

        hipLaunchKernelGGL(rmsnorm_f16, dim3(MV), dim3(256), 0, stream, xv, vlm_ln2 + (long)i * WVd, h2v, WVd);
        hipLaunchKernelGGL(rmsnorm_f16, dim3(ME), dim3(256), 0, stream, xe, exp_ln2 + (long)i * WEd, h2e, WEd);

        gemmW(h2v, vlm_gw + (long)i * FVd * WVd, gv, MV, FVd, WVd);
        gemmW(h2v, vlm_uw + (long)i * FVd * WVd, uv, MV, FVd, WVd);
        hipLaunchKernelGGL(gelu_mul_f16, launch1d((long)MV * FVd), dim3(256), 0, stream,
                           gv, uv, guv, (long)MV * FVd);
        gemmW(guv, vlm_dw + (long)i * WVd * FVd, dv, MV, WVd, FVd);
        hipLaunchKernelGGL(add_inplace, launch1d(nXV), dim3(256), 0, stream, xv, dv, nXV);

        gemmW(h2e, exp_gw + (long)i * FEd * WEd, ge, ME, FEd, WEd);
        gemmW(h2e, exp_uw + (long)i * FEd * WEd, ue, ME, FEd, WEd);
        hipLaunchKernelGGL(gelu_mul_f16, launch1d((long)ME * FEd), dim3(256), 0, stream,
                           ge, ue, gue, (long)ME * FEd);
        gemmW(gue, exp_dw + (long)i * WEd * FEd, de, ME, WEd, FEd);
        hipLaunchKernelGGL(add_inplace, launch1d(nXE), dim3(256), 0, stream, xe, de, nXE);
    }

    // outputs: (vlm_x, exp_x) concatenated
    float* out = (float*)d_out;
    (void)hipMemcpyAsync(out,       xv, (size_t)nXV * 4, hipMemcpyDeviceToDevice, stream);
    (void)hipMemcpyAsync(out + nXV, xe, (size_t)nXE * 4, hipMemcpyDeviceToDevice, stream);
}
